// Head_61254823576141
// MI455X (gfx1250) — matrix operand order repaired and checked by reading, not compiled, not run on hardware
//
#include <hip/hip_runtime.h>
#include <hip/hip_bf16.h>

// Problem sizes (match reference)
#define N_EMBD 2048
#define SEQ    2048
#define HS     128
#define BATCH  4
#define KC     32            // K-chunk per WMMA step
#define NCHUNK (N_EMBD / KC)
#define LSTR   40            // padded LDS row stride (elements): 80B, 16B-aligned, conflict-free

typedef __bf16 bf16_t;
typedef __attribute__((ext_vector_type(8)))  bf16_t v8bf;
typedef __attribute__((ext_vector_type(16))) bf16_t v16bf;
typedef __attribute__((ext_vector_type(8)))  float  v8f;

// Build a 16-element bf16 fragment from two 16-byte chunks (two b128 loads).
static __device__ __forceinline__ v16bf load_frag(const bf16_t* lo, const bf16_t* hi) {
  v8bf a = *(const v8bf*)lo;
  v8bf b = *(const v8bf*)hi;
  v16bf r;
#pragma unroll
  for (int i = 0; i < 8; ++i) { r[i] = a[i]; r[i + 8] = b[i]; }
  return r;
}

static __device__ __forceinline__ v8f wmma_bf16(v16bf a, v16bf b, v8f c) {
  // D = A(16x32 bf16) * B(32x16 bf16) + C(16x16 f32)
  return __builtin_amdgcn_wmma_f32_16x16x32_bf16(false, a, false, b, (short)0, c,
                                                 false, false);
}

// Async global->LDS copy of 16 bytes per lane (CDNA5, tracked by ASYNCcnt).
// Generic shared pointers on amdgcn carry the LDS byte offset in the low 32 bits.
static __device__ __forceinline__ void async_ld16(unsigned lds_off, const bf16_t* g) {
  unsigned long long ga = (unsigned long long)g;
  asm volatile("global_load_async_to_lds_b128 %0, %1, off"
               :: "v"(lds_off), "v"(ga) : "memory");
}

template <int N>
static __device__ __forceinline__ void wait_async() {
  asm volatile("s_wait_asynccnt %0" :: "n"(N) : "memory");
}

// ---------------------------------------------------------------------------
// Kernel 0a: x f32 -> bf16 (memory bound; 64MB read + 32MB write @ 23.3TB/s).
// ---------------------------------------------------------------------------
__global__ __launch_bounds__(256) void cvt_x_kernel(const float* __restrict__ x,
                                                    bf16_t* __restrict__ xb) {
  const size_t i = ((size_t)blockIdx.x * 256 + threadIdx.x) * 8;
  float4 a = *(const float4*)(x + i);
  float4 b = *(const float4*)(x + i + 4);
  v8bf o;
  o[0] = (bf16_t)a.x; o[1] = (bf16_t)a.y; o[2] = (bf16_t)a.z; o[3] = (bf16_t)a.w;
  o[4] = (bf16_t)b.x; o[5] = (bf16_t)b.y; o[6] = (bf16_t)b.z; o[7] = (bf16_t)b.w;
  *(v8bf*)(xb + i) = o;
}

// ---------------------------------------------------------------------------
// Kernel 0b: W f32 [C][H] -> bf16 transposed [3][H][C] so B-fragments are
// contiguous 16-element runs in global memory (no LDS staging needed for W).
// ---------------------------------------------------------------------------
__global__ __launch_bounds__(256) void cvt_w_kernel(
    const float* __restrict__ wq, const float* __restrict__ wk,
    const float* __restrict__ wv, bf16_t* __restrict__ wtg) {
  const int idx   = blockIdx.x * 256 + threadIdx.x;    // 3 * 2048 * 8 = 49152
  const int which = idx / (N_EMBD * 8);
  const int rem   = idx % (N_EMBD * 8);
  const int c     = rem >> 3;
  const int n0    = (rem & 7) * 16;
  const float* w  = (which == 0) ? wq : (which == 1) ? wk : wv;
  const float* src = w + (size_t)c * HS + n0;
  bf16_t* dst = wtg + (size_t)which * HS * N_EMBD;
#pragma unroll
  for (int i = 0; i < 16; i += 4) {
    float4 f = *(const float4*)(src + i);
    dst[(size_t)(n0 + i + 0) * N_EMBD + c] = (bf16_t)f.x;
    dst[(size_t)(n0 + i + 1) * N_EMBD + c] = (bf16_t)f.y;
    dst[(size_t)(n0 + i + 2) * N_EMBD + c] = (bf16_t)f.z;
    dst[(size_t)(n0 + i + 3) * N_EMBD + c] = (bf16_t)f.w;
  }
}

// ---------------------------------------------------------------------------
// Kernel 1: QKV projection.  q/k -> bf16 [B][T][H]; v -> bf16 [B][H][T].
// Block: 256 threads (8 waves), one of {Q,K,V} per blockIdx.y, 128x128 tile.
// x tile is double-buffered in LDS via global_load_async_to_lds_b128 with
// partial s_wait_asynccnt; W B-frags load straight from global (L2-resident).
// Inner loop preloads all 8 A-frags, then issues 8 back-to-back WMMAs.
// ---------------------------------------------------------------------------
__global__ __launch_bounds__(256) void qkv_proj_kernel(
    const bf16_t* __restrict__ xb, const bf16_t* __restrict__ wtg,
    const float* __restrict__ bq, const float* __restrict__ bk,
    const float* __restrict__ bv,
    bf16_t* __restrict__ qb, bf16_t* __restrict__ kbm, bf16_t* __restrict__ vtm)
{
  __shared__ bf16_t xs[2][128][LSTR];   // double-buffered bf16 x tile

  const int tid  = threadIdx.x;
  const int lane = tid & 31;
  const int wave = tid >> 5;
  const int half = lane >> 4;
  const int l16  = lane & 15;

  const int bidx  = blockIdx.z;
  const int which = blockIdx.y;          // 0=Q 1=K 2=V
  const int row0  = blockIdx.x * 128;

  const float* bias = (which == 0) ? bq : (which == 1) ? bk : bv;

  const int n   = wave * 16 + l16;       // this lane's output column
  const int k0a = half * 16;             // A-frag K base (contiguous, matches B)
  const int k0b = half * 16;             // B-frag K base (contiguous, matches A)

  const bf16_t* xblk = xb + ((size_t)bidx * SEQ + row0) * N_EMBD;
  const bf16_t* wrow = wtg + ((size_t)which * HS + n) * N_EMBD;

  // async staging coordinates: 128 rows x 32 cols bf16 = 512 x 16B chunks
  const int srow = tid >> 1;
  const int scol = (tid & 1) * 16;

  v8f acc[8];
#pragma unroll
  for (int m = 0; m < 8; ++m)
#pragma unroll
    for (int r = 0; r < 8; ++r) acc[m][r] = 0.0f;

  // prologue: stage chunk 0 into buffer 0
  {
    const bf16_t* g = xblk + (size_t)srow * N_EMBD + scol;
    unsigned l = (unsigned)(unsigned long long)&xs[0][srow][scol];
    async_ld16(l, g);
    async_ld16(l + 16, g + 8);
  }

  for (int i = 0; i < NCHUNK; ++i) {
    const int kk  = i * KC;
    const int buf = i & 1;

    if (i + 1 < NCHUNK) {
      // prefetch next chunk into the other buffer (safe: it was last read in
      // iteration i-1 and the trailing barrier of that iteration has passed)
      const bf16_t* g = xblk + (size_t)srow * N_EMBD + (kk + KC) + scol;
      unsigned l = (unsigned)(unsigned long long)&xs[buf ^ 1][srow][scol];
      async_ld16(l, g);
      async_ld16(l + 16, g + 8);
      wait_async<2>();     // asyncs complete in order: current buffer landed
    } else {
      wait_async<0>();
    }
    __syncthreads();

    // B-fragment straight from global (contiguous 16 K values, L2-resident)
    v16bf bfrag = load_frag(wrow + kk + k0b, wrow + kk + k0b + 8);

    // preload all 8 A-frags, then a back-to-back WMMA chain
    v16bf af[8];
#pragma unroll
    for (int m = 0; m < 8; ++m) {
      const bf16_t* ar = &xs[buf][m * 16 + l16][0];
      af[m] = load_frag(ar + k0a, ar + k0a + 8);
    }
#pragma unroll
    for (int m = 0; m < 8; ++m)
      acc[m] = wmma_bf16(af[m], bfrag, acc[m]);

    __syncthreads();   // protect buffer reuse two iterations out
  }

  const float bn = bias[n];
#pragma unroll
  for (int m = 0; m < 8; ++m) {
#pragma unroll
    for (int r = 0; r < 8; ++r) {
      const int row = row0 + m * 16 + r + half * 8;          // C-layout row
      const float val = acc[m][r] + bn;
      if (which == 0)      qb [((size_t)bidx * SEQ + row) * HS + n] = (bf16_t)val;
      else if (which == 1) kbm[((size_t)bidx * SEQ + row) * HS + n] = (bf16_t)val;
      else                 vtm[((size_t)bidx * HS + n) * SEQ + row] = (bf16_t)val;
    }
  }
}

// ---------------------------------------------------------------------------
// Kernel 2: causal flash attention.  Each wave owns 16 query rows; streams
// 32-key blocks up to its causal limit (no block barriers -> waves diverge
// freely). K/V B-frags load straight from global (layouts pre-arranged).
// ---------------------------------------------------------------------------
__global__ __launch_bounds__(256) void flash_attn_kernel(
    const bf16_t* __restrict__ qb, const bf16_t* __restrict__ kbm,
    const bf16_t* __restrict__ vtm, float* __restrict__ out)
{
  __shared__ bf16_t pLds[8][16][LSTR];   // per-wave P transpose buffer

  const int tid  = threadIdx.x;
  const int lane = tid & 31;
  const int wave = tid >> 5;
  const int half = lane >> 4;
  const int l16  = lane & 15;

  const int bidx  = blockIdx.y;
  const int qbase = blockIdx.x * 128 + wave * 16;

  const bf16_t* qp = qb  + ((size_t)bidx * SEQ + qbase) * HS;
  const bf16_t* kp = kbm + (size_t)bidx * SEQ * HS;
  const bf16_t* vp = vtm + (size_t)bidx * HS * SEQ;

  const int k0a = half * 16;
  const int k0b = half * 16;

  // Q A-fragments for the whole head dim (4 chunks of K=32), kept resident.
  v16bf qf[4];
#pragma unroll
  for (int c = 0; c < 4; ++c) {
    const bf16_t* p = qp + (size_t)l16 * HS + c * 32;
    qf[c] = load_frag(p + k0a, p + k0a + 8);
  }

  v8f o[8];
#pragma unroll
  for (int nt = 0; nt < 8; ++nt)
#pragma unroll
    for (int r = 0; r < 8; ++r) o[nt][r] = 0.0f;

  float m_i[8], l_i[8];
#pragma unroll
  for (int r = 0; r < 8; ++r) { m_i[r] = -__builtin_inff(); l_i[r] = 0.0f; }

  const float scale = 0.02209708691207961f;  // 1/sqrt(N_EMBD) per reference
  const int qmax = qbase + 15;

  for (int kb = 0; kb <= qmax; kb += 32) {
    // ---- S = Q K^T for two 16-key tiles ----
    v8f s[2];
#pragma unroll
    for (int t = 0; t < 2; ++t) {
      v8f sa;
#pragma unroll
      for (int r = 0; r < 8; ++r) sa[r] = 0.0f;
      const bf16_t* kr = kp + (size_t)(kb + t * 16 + l16) * HS;  // lane = key col
#pragma unroll
      for (int c = 0; c < 4; ++c) {
        v16bf bf = load_frag(kr + c * 32 + k0b, kr + c * 32 + k0b + 8);
        sa = wmma_bf16(qf[c], bf, sa);
      }
      s[t] = sa;
    }

    // ---- scale, causal mask, online softmax (per row) ----
    float alpha[8];
#pragma unroll
    for (int r = 0; r < 8; ++r) {
      const int row  = qbase + r + half * 8;
      const int key0 = kb + l16;
      float s0 = (key0      <= row) ? s[0][r] * scale : -__builtin_inff();
      float s1 = (key0 + 16 <= row) ? s[1][r] * scale : -__builtin_inff();
      float mx = fmaxf(s0, s1);
#pragma unroll
      for (int d = 1; d < 16; d <<= 1) mx = fmaxf(mx, __shfl_xor(mx, d, 32));
      const float mnew = fmaxf(m_i[r], mx);
      const float p0 = __expf(s0 - mnew);
      const float p1 = __expf(s1 - mnew);
      float rs = p0 + p1;
#pragma unroll
      for (int d = 1; d < 16; d <<= 1) rs += __shfl_xor(rs, d, 32);
      alpha[r] = __expf(m_i[r] - mnew);
      l_i[r]   = l_i[r] * alpha[r] + rs;
      m_i[r]   = mnew;
      // park P (C-layout) into LDS for re-read in A-layout
      pLds[wave][r + half * 8][l16]      = (bf16_t)p0;
      pLds[wave][r + half * 8][16 + l16] = (bf16_t)p1;
    }

    // per-wave LDS visibility: wave is lockstep, just drain DS counter
    asm volatile("s_wait_dscnt 0" ::: "memory");

    const bf16_t* pr = &pLds[wave][l16][0];
    v16bf pf = load_frag(pr + k0a, pr + k0a + 8);

    // ---- O = O*alpha + P V  (8 head-dim tiles) ----
#pragma unroll
    for (int nt = 0; nt < 8; ++nt) {
      v8f on = o[nt];
#pragma unroll
      for (int r = 0; r < 8; ++r) on[r] *= alpha[r];
      const bf16_t* vr = vp + (size_t)(nt * 16 + l16) * SEQ + kb + k0b;
      v16bf vf = load_frag(vr, vr + 8);
      o[nt] = wmma_bf16(pf, vf, on);
    }
  }

  // ---- epilogue: normalize and store f32 ----
  float* op = out + ((size_t)bidx * SEQ + qbase) * HS;
#pragma unroll
  for (int nt = 0; nt < 8; ++nt) {
#pragma unroll
    for (int r = 0; r < 8; ++r) {
      const int row = r + half * 8;
      const int col = nt * 16 + l16;
      op[(size_t)row * HS + col] = o[nt][r] / l_i[r];
    }
  }
}

// ---------------------------------------------------------------------------
extern "C" void kernel_launch(void* const* d_in, const int* in_sizes, int n_in,
                              void* d_out, int out_size, void* d_ws, size_t ws_size,
                              hipStream_t stream) {
  (void)in_sizes; (void)n_in; (void)out_size; (void)ws_size;

  const float* x  = (const float*)d_in[0];
  const float* wq = (const float*)d_in[1];
  const float* bq = (const float*)d_in[2];
  const float* wk = (const float*)d_in[3];
  const float* bk = (const float*)d_in[4];
  const float* wv = (const float*)d_in[5];
  const float* bv = (const float*)d_in[6];

  // Workspace layout (bf16): Q[B][T][H], K[B][T][H], V^T[B][H][T],
  // x[B][T][C], W^T[3][H][C]  -> ~40 MB total.
  bf16_t* qb  = (bf16_t*)d_ws;
  bf16_t* kbm = qb  + (size_t)BATCH * SEQ * HS;
  bf16_t* vtm = kbm + (size_t)BATCH * SEQ * HS;
  bf16_t* xbb = vtm + (size_t)BATCH * SEQ * HS;
  bf16_t* wtg = xbb + (size_t)BATCH * SEQ * N_EMBD;

  const size_t xelems = (size_t)BATCH * SEQ * N_EMBD;      // 16,777,216
  cvt_x_kernel<<<dim3((unsigned)(xelems / (256 * 8))), 256, 0, stream>>>(x, xbb);

  cvt_w_kernel<<<dim3(3 * N_EMBD * 8 / 256), 256, 0, stream>>>(wq, wk, wv, wtg);

  dim3 gA(SEQ / 128, 3, BATCH);
  qkv_proj_kernel<<<gA, 256, 0, stream>>>(xbb, wtg, bq, bk, bv, qb, kbm, vtm);

  dim3 gB(SEQ / 128, BATCH);
  flash_attn_kernel<<<gB, 256, 0, stream>>>(qb, kbm, vtm, (float*)d_out);
}
